// ProteinGrid_77412490543539
// MI455X (gfx1250) — matrix operand order repaired and checked by reading, not compiled, not run on hardware
//
#include <hip/hip_runtime.h>
#include <hip/hip_bf16.h>

#define HDIM 128
#define EPSC 1e-5f

typedef __attribute__((ext_vector_type(16))) _Float16 v16h;
typedef __attribute__((ext_vector_type(8)))  _Float16 v8h;
typedef __attribute__((ext_vector_type(8)))  float    v8f;

// ---------------------------------------------------------------------------
// WMMA fragment helpers (wave32, 16x16x32 f16 -> f32)
// A layout used here (16x32 f16): lanes 0-15 row=lane, K 0..15; lanes 16-31 row=lane-16, K 16..31
// ---------------------------------------------------------------------------
__device__ __forceinline__ v16h frag_a_lds(const _Float16* tile, int stride, int lane) {
    const int row  = lane & 15;
    const int koff = (lane >> 4) << 4;            // 0 or 16
    const _Float16* p = tile + row * stride + koff;
    v8h lo = *(const v8h*)(p);                    // K = koff .. koff+7
    v8h hi = *(const v8h*)(p + 8);                // K = koff+8 .. koff+15
    v16h r;
#pragma unroll
    for (int i = 0; i < 8; ++i) { r[i] = lo[i]; r[i + 8] = hi[i]; }
    return r;
}

// B fragments are pre-swizzled: [(kb*NT + nt)*32 + lane] * 16 halves, one 32B load
__device__ __forceinline__ v16h frag_b_sw(const _Float16* wsw, int kb, int NT, int nt, int lane) {
    return *(const v16h*)(wsw + ((((kb * NT + nt) * 32) + lane) << 4));
}

__device__ __forceinline__ v8f wmma_acc(v16h a, v16h b, v8f c) {
    return __builtin_amdgcn_wmma_f32_16x16x32_f16(false, a, false, b, (short)0, c, false, false);
}

__device__ __forceinline__ v8f mm_tile(const _Float16* atile, int stride, int KB,
                                       const _Float16* wsw, int NT, int nt, int lane) {
    v8f acc = {};
#pragma unroll
    for (int kb = 0; kb < KB; ++kb) {
        v16h a = frag_a_lds(atile + kb * 32, stride, lane);
        v16h b = frag_b_sw(wsw, kb, NT, nt, lane);
        acc = wmma_acc(a, b, acc);
    }
    return acc;
}

// fast SiLU: one v_exp_f32 + one v_rcp_f32 (no IEEE div chain)
__device__ __forceinline__ float silu_f(float x) {
    return x * __builtin_amdgcn_rcpf(1.f + __expf(-x));
}

// C/D layout: lane L -> col = L&15 ; VGPR r -> row = r + 8*(L>=16)
__device__ __forceinline__ void store_tile_lds(_Float16* dst, int stride, v8f acc,
                                               const float* __restrict__ bias,
                                               int nt, int lane, bool dosilu) {
    const int col   = (nt << 4) + (lane & 15);
    const int rbase = (lane >> 4) << 3;
    const float bv  = bias[col];
#pragma unroll
    for (int r = 0; r < 8; ++r) {
        float v = acc[r] + bv;
        if (dosilu) v = silu_f(v);
        dst[(rbase + r) * stride + col] = (_Float16)v;
    }
}

// ---------------------------------------------------------------------------
// Weight pre-swizzle: fp32 [K][N] row-major -> fp16 B-fragment layout.
// B layout: lanes 0-15 K=0..15 (2/VGPR), lanes 16-31 K=16..31.
// ---------------------------------------------------------------------------
__global__ void pg_swizzle_mlp(const float* __restrict__ W, int Kpad, int Ksrc, int N,
                               const float* __restrict__ colscale, _Float16* __restrict__ out) {
    int idx = blockIdx.x * blockDim.x + threadIdx.x;
    int total = Kpad * N;
    if (idx >= total) return;
    int j    = idx & 15;
    int lane = (idx >> 4) & 31;
    int t    = idx >> 9;
    int NT   = N >> 4;
    int nt   = t % NT;
    int kb   = t / NT;
    int v = j >> 1, h = j & 1;
    int klocal = 2 * v + h + (((lane >> 4) & 1) << 4);
    int k = kb * 32 + klocal;
    int n = (nt << 4) + (lane & 15);
    float val = (k < Ksrc) ? W[k * N + n] : 0.f;
    if (colscale) val *= colscale[n] * rsqrtf(1.f + EPSC);
    out[idx] = (_Float16)val;
}

// conv weights: fp32 OIDHW [O][I][T] -> fp16 [tap][kb][ntile][lane][16], BN gamma folded
__global__ void pg_swizzle_conv(const float* __restrict__ W, int O, int I, int T,
                                const float* __restrict__ gamma, _Float16* __restrict__ out) {
    int idx = blockIdx.x * blockDim.x + threadIdx.x;
    int total = T * I * O;
    if (idx >= total) return;
    int j    = idx & 15;
    int lane = (idx >> 4) & 31;
    int rest = idx >> 9;
    int NT = O >> 4, KB = I >> 5;
    int nt  = rest % NT; rest /= NT;
    int kb  = rest % KB;
    int tap = rest / KB;
    int v = j >> 1, h = j & 1;
    int klocal = 2 * v + h + (((lane >> 4) & 1) << 4);
    int i = kb * 32 + klocal;
    int o = (nt << 4) + (lane & 15);
    float val = W[(o * I + i) * T + tap] * gamma[o] * rsqrtf(1.f + EPSC);
    out[idx] = (_Float16)val;
}

__global__ void pg_fold_bias(const float* __restrict__ b2, const float* __restrict__ gamma,
                             const float* __restrict__ beta, float* __restrict__ out) {
    int c = threadIdx.x;
    float s = gamma[c] * rsqrtf(1.f + EPSC);
    out[c] = b2[c] * s + beta[c];
}

// ---------------------------------------------------------------------------
// Fused edge pipeline: 16 edges / block -> mean over 8-edge groups -> agg (fp16)
// ---------------------------------------------------------------------------
__global__ __launch_bounds__(256) void pg_edge_kernel(
    const float* __restrict__ node_emb, const float* __restrict__ node_pos,
    const float* __restrict__ grid_pos, const int* __restrict__ edge_row,
    const int* __restrict__ edge_col,
    const _Float16* __restrict__ w_e1, const float* __restrict__ b_e1,
    const _Float16* __restrict__ w_e2, const float* __restrict__ b_e2,
    const _Float16* __restrict__ w_m1, const float* __restrict__ b_m1,
    const _Float16* __restrict__ w_m2, const float* __restrict__ b_m2,
    _Float16* __restrict__ aggbuf) {
    __shared__ __align__(16) _Float16 s_attr[16 * 32];
    __shared__ __align__(16) _Float16 s_h[16 * HDIM];
    __shared__ __align__(16) _Float16 s_min[16 * 256];   // [:128]=node emb, [128:]=pos_emb
    __shared__ __align__(16) _Float16 s_msg[16 * HDIM];
    __shared__ int s_row[16];

    const int tid  = threadIdx.x;
    const int lane = tid & 31;
    const int wv   = tid >> 5;          // 0..7 = n-tile
    const int e0   = blockIdx.x * 16;

    if (tid < 16) s_row[tid] = edge_row[e0 + tid];
    __syncthreads();

    // stage edge_attr [16][6] zero-padded to K=32 (fp16)
    {
        int e  = tid >> 4;
        int k2 = (tid & 15) * 2;
        int cc = edge_col[e0 + e];
        int rr = s_row[e];
        float v0 = (k2 < 3) ? node_pos[rr * 3 + k2] : ((k2 < 6) ? grid_pos[cc * 3 + k2 - 3] : 0.f);
        int k3 = k2 + 1;
        float v1 = (k3 < 3) ? node_pos[rr * 3 + k3] : ((k3 < 6) ? grid_pos[cc * 3 + k3 - 3] : 0.f);
        s_attr[e * 32 + k2]     = (_Float16)v0;
        s_attr[e * 32 + k2 + 1] = (_Float16)v1;
    }
    // stage gathered node embeddings into msg_in[:, :128]
    for (int i = tid; i < 16 * HDIM; i += 256) {
        int e = i >> 7, c = i & 127;
        s_min[e * 256 + c] = (_Float16)node_emb[s_row[e] * HDIM + c];
    }
    __syncthreads();

    // edge MLP layer 1 (K=32 padded), SiLU
    v8f acc = mm_tile(s_attr, 32, 1, w_e1, 8, wv, lane);
    store_tile_lds(s_h, HDIM, acc, b_e1, wv, lane, true);
    __syncthreads();
    // edge MLP layer 2 (K=128) -> pos_emb = msg_in[:, 128:]
    acc = mm_tile(s_h, HDIM, 4, w_e2, 8, wv, lane);
    store_tile_lds(s_min + 128, 256, acc, b_e2, wv, lane, false);
    __syncthreads();
    // message MLP layer 1 (K=256), SiLU
    acc = mm_tile(s_min, 256, 8, w_m1, 8, wv, lane);
    store_tile_lds(s_h, HDIM, acc, b_m1, wv, lane, true);
    __syncthreads();
    // message MLP layer 2 (K=128)
    acc = mm_tile(s_h, HDIM, 4, w_m2, 8, wv, lane);
    store_tile_lds(s_msg, HDIM, acc, b_m2, wv, lane, false);
    __syncthreads();

    // scatter-mean: edges are contiguous groups of 8 per grid point
    {
        int g = tid >> 7;          // 0..1 grid points in this tile
        int c = tid & 127;
        float s = 0.f;
#pragma unroll
        for (int r = 0; r < 8; ++r) s += (float)s_msg[(g * 8 + r) * HDIM + c];
        aggbuf[(blockIdx.x * 2 + g) * HDIM + c] = (_Float16)(s * 0.125f);
    }
}

// ---------------------------------------------------------------------------
// Update MLP + input BN fold -> channels-last fp16 grid [b][z][y][x][c]
// ---------------------------------------------------------------------------
__global__ __launch_bounds__(256) void pg_update_kernel(
    const _Float16* __restrict__ aggbuf,
    const _Float16* __restrict__ w_u1, const float* __restrict__ b_u1,
    const _Float16* __restrict__ w_u2, const float* __restrict__ b_fold,
    _Float16* __restrict__ act) {
    __shared__ __align__(16) _Float16 s_a[16 * HDIM];
    __shared__ __align__(16) _Float16 s_h[16 * HDIM];
    const int tid = threadIdx.x, lane = tid & 31, wv = tid >> 5;
    const int g0 = blockIdx.x * 16;
    for (int i = tid; i < 16 * HDIM; i += 256) s_a[i] = aggbuf[g0 * HDIM + i];
    __syncthreads();
    v8f acc = mm_tile(s_a, HDIM, 4, w_u1, 8, wv, lane);
    store_tile_lds(s_h, HDIM, acc, b_u1, wv, lane, true);
    __syncthreads();
    acc = mm_tile(s_h, HDIM, 4, w_u2, 8, wv, lane);
    const int col   = (wv << 4) + (lane & 15);
    const int rbase = (lane >> 4) << 3;
    const float bv  = b_fold[col];
#pragma unroll
    for (int r = 0; r < 8; ++r)
        act[(g0 + rbase + r) * HDIM + col] = (_Float16)(acc[r] + bv);
}

// ---------------------------------------------------------------------------
// Implicit-GEMM 3D conv: block = (b, z, 4-row y-tile) -> M=64 voxels x 128 out ch.
// Per dz: stage the union of input rows [y0-PAD .. y0+3+PAD] as padded x-rows in
// LDS once; each wave keeps 4 accumulators and reuses every weight (B) fragment
// across the 4 y-rows from registers -> 4x less weight traffic, 2 barriers/dz.
// ---------------------------------------------------------------------------
template <int KS, bool RELU, bool ADD>
__global__ __launch_bounds__(256) void pg_conv_kernel(
    const _Float16* __restrict__ act, const _Float16* __restrict__ wsw,
    const float* __restrict__ bias, const _Float16* __restrict__ addsrc,
    _Float16* __restrict__ out) {
    constexpr int PAD = KS / 2;
    constexpr int XP  = 16 + 2 * PAD;      // padded x extent
    constexpr int YT  = 4;                 // output y rows per block
    constexpr int YR  = YT + 2 * PAD;      // staged input rows per dz
    __shared__ __align__(16) _Float16 s_a[YR * XP * HDIM];
    const int tid = threadIdx.x, lane = tid & 31, wv = tid >> 5;
    const int bi = blockIdx.x;
    const int y0 = (bi & 3) * YT;
    const int z  = (bi >> 2) & 15;
    const int b  = bi >> 6;

    v8f acc[YT] = {};
#pragma unroll 1
    for (int dz = 0; dz < KS; ++dz) {
        const int zz = z + dz - PAD;
        const bool zok = (unsigned)zz < 16u;
        __syncthreads();
        // stage YR padded x-rows (zero-filled borders), 16B granules
        for (int idx = tid; idx < YR * XP * 16; idx += 256) {
            const int ryp = idx / (XP * 16);
            const int rem = idx - ryp * (XP * 16);
            const int xp  = rem >> 4;
            const int seg = rem & 15;
            const int yy  = y0 + ryp - PAD;
            const int xx  = xp - PAD;
            v8h v = {};
            if (zok && ((unsigned)yy < 16u) && ((unsigned)xx < 16u))
                v = *(const v8h*)(act + ((((b * 16 + zz) * 16 + yy) * 16 + xx) * HDIM) + seg * 8);
            *(v8h*)(s_a + (ryp * XP + xp) * HDIM + seg * 8) = v;
        }
        __syncthreads();
#pragma unroll 1
        for (int dy = 0; dy < KS; ++dy) {
#pragma unroll
            for (int dx = 0; dx < KS; ++dx) {
                const int tap = (dz * KS + dy) * KS + dx;
#pragma unroll
                for (int c = 0; c < 4; ++c) {
                    const _Float16* wp =
                        wsw + ((size_t)((tap * 4 + c) * 8 + wv) * 32 + lane) * 16;
                    __builtin_prefetch(wp + 8 * 32 * 16, 0, 1);
                    const v16h bf = *(const v16h*)wp;   // reused for all 4 y rows
#pragma unroll
                    for (int ry = 0; ry < YT; ++ry) {
                        v16h a = frag_a_lds(s_a + ((ry + dy) * XP + dx) * HDIM + c * 32,
                                            HDIM, lane);
                        acc[ry] = wmma_acc(a, bf, acc[ry]);
                    }
                }
            }
        }
    }
    const int col   = (wv << 4) + (lane & 15);
    const int rbase = (lane >> 4) << 3;
    const float bv  = bias[col];
#pragma unroll
    for (int ry = 0; ry < YT; ++ry) {
#pragma unroll
        for (int r = 0; r < 8; ++r) {
            const int vox = ((b * 16 + z) * 16 + (y0 + ry)) * 16 + rbase + r;
            float v = acc[ry][r] + bv;
            if (ADD)  v += (float)addsrc[vox * HDIM + col];
            if (RELU) v = fmaxf(v, 0.f);
            out[vox * HDIM + col] = (_Float16)v;
        }
    }
}

// ---------------------------------------------------------------------------
// Global max pool over 4096 voxels per (batch, channel)
// ---------------------------------------------------------------------------
__global__ __launch_bounds__(256) void pg_pool_kernel(const _Float16* __restrict__ act,
                                                      float* __restrict__ pooled) {
    __shared__ float s_m[256];
    const int b = blockIdx.x;
    const int c = threadIdx.x & 127;
    const int half = threadIdx.x >> 7;                 // split voxel range in two
    float m = -3.0e38f;
    for (int v = half * 2048; v < (half + 1) * 2048; ++v)
        m = fmaxf(m, (float)act[(b * 4096 + v) * HDIM + c]);
    s_m[threadIdx.x] = m;
    __syncthreads();
    if (half == 0) pooled[b * HDIM + c] = fmaxf(s_m[c], s_m[c + 128]);
}

__global__ void pg_head_kernel(const float* __restrict__ pooled, const float* __restrict__ fc_w,
                               const float* __restrict__ fc_b, float* __restrict__ out) {
    __shared__ float sl[2][20];
    const int tid = threadIdx.x;
    if (tid < 40) {
        const int b = tid / 20, j = tid % 20;
        float a = fc_b[j];
        for (int c = 0; c < HDIM; ++c) a += pooled[b * HDIM + c] * fc_w[c * 20 + j];
        sl[b][j] = a;
    }
    __syncthreads();
    if (tid < 2) {
        float mx = -3.0e38f;
        for (int j = 0; j < 20; ++j) mx = fmaxf(mx, sl[tid][j]);
        float s = 0.f;
        for (int j = 0; j < 20; ++j) s += __expf(sl[tid][j] - mx);
        const float lse = mx + __logf(s);
        for (int j = 0; j < 20; ++j) out[tid * 20 + j] = sl[tid][j] - lse;
    }
}

// ---------------------------------------------------------------------------
extern "C" void kernel_launch(void* const* d_in, const int* in_sizes, int n_in,
                              void* d_out, int out_size, void* d_ws, size_t ws_size,
                              hipStream_t stream) {
    const float* node_emb = (const float*)d_in[0];
    const float* node_pos = (const float*)d_in[1];
    const float* grid_pos = (const float*)d_in[2];
    const int*   edge_row = (const int*)d_in[3];
    const int*   edge_col = (const int*)d_in[4];
    const float* edge_w1 = (const float*)d_in[5];
    const float* edge_b1 = (const float*)d_in[6];
    const float* edge_w2 = (const float*)d_in[7];
    const float* edge_b2 = (const float*)d_in[8];
    const float* msg_w1  = (const float*)d_in[9];
    const float* msg_b1  = (const float*)d_in[10];
    const float* msg_w2  = (const float*)d_in[11];
    const float* msg_b2  = (const float*)d_in[12];
    const float* upd_w1  = (const float*)d_in[13];
    const float* upd_b1  = (const float*)d_in[14];
    const float* upd_w2  = (const float*)d_in[15];
    const float* upd_b2  = (const float*)d_in[16];
    const float* in_gamma = (const float*)d_in[17];
    const float* in_beta  = (const float*)d_in[18];
    const float* blk_conv1 = (const float*)d_in[19];
    const float* blk_bn1_g = (const float*)d_in[20];
    const float* blk_bn1_b = (const float*)d_in[21];
    const float* blk_conv2 = (const float*)d_in[22];
    const float* blk_bn2_g = (const float*)d_in[23];
    const float* blk_bn2_b = (const float*)d_in[24];
    const float* blk_conv5 = (const float*)d_in[25];
    const float* blk_bns_g = (const float*)d_in[26];
    const float* blk_bns_b = (const float*)d_in[27];
    const float* fc_w = (const float*)d_in[28];
    const float* fc_b = (const float*)d_in[29];
    float* out = (float*)d_out;

    // --- workspace carve-up (all offsets 256B aligned) ---
    char* wsb = (char*)d_ws;
    size_t off = 0;
    auto take = [&](size_t bytes) -> char* {
        char* p = wsb + off;
        off += (bytes + 255) & ~(size_t)255;
        return p;
    };
    const size_t GRIDN = 8192;                         // B*G^3
    _Float16* aggbuf = (_Float16*)take(GRIDN * HDIM * 2);
    _Float16* act_a  = (_Float16*)take(GRIDN * HDIM * 2);
    _Float16* act_b  = (_Float16*)take(GRIDN * HDIM * 2);
    _Float16* tmp1   = (_Float16*)take(GRIDN * HDIM * 2);
    _Float16* tmp2   = (_Float16*)take(GRIDN * HDIM * 2);
    _Float16* sw_e1  = (_Float16*)take(32 * 128 * 2);
    _Float16* sw_e2  = (_Float16*)take(128 * 128 * 2);
    _Float16* sw_m1  = (_Float16*)take(256 * 128 * 2);
    _Float16* sw_m2  = (_Float16*)take(128 * 128 * 2);
    _Float16* sw_u1  = (_Float16*)take(128 * 128 * 2);
    _Float16* sw_u2  = (_Float16*)take(128 * 128 * 2);
    _Float16* sw_c1[3], *sw_c2[3], *sw_c5[3];
    for (int i = 0; i < 3; ++i) sw_c1[i] = (_Float16*)take(27 * 128 * 128 * 2);
    for (int i = 0; i < 3; ++i) sw_c2[i] = (_Float16*)take(27 * 128 * 128 * 2);
    for (int i = 0; i < 3; ++i) sw_c5[i] = (_Float16*)take(125 * 128 * 128 * 2);
    float* b_fold = (float*)take(128 * 4);
    float* pooled = (float*)take(2 * 128 * 4);

    // --- weight conversion / swizzle (+BN fold) ---
    auto gridFor = [](int total) { return dim3((total + 255) / 256); };
    pg_swizzle_mlp<<<gridFor(32 * 128), 256, 0, stream>>>(edge_w1, 32, 6, 128, nullptr, sw_e1);
    pg_swizzle_mlp<<<gridFor(128 * 128), 256, 0, stream>>>(edge_w2, 128, 128, 128, nullptr, sw_e2);
    pg_swizzle_mlp<<<gridFor(256 * 128), 256, 0, stream>>>(msg_w1, 256, 256, 128, nullptr, sw_m1);
    pg_swizzle_mlp<<<gridFor(128 * 128), 256, 0, stream>>>(msg_w2, 128, 128, 128, nullptr, sw_m2);
    pg_swizzle_mlp<<<gridFor(128 * 128), 256, 0, stream>>>(upd_w1, 128, 128, 128, nullptr, sw_u1);
    pg_swizzle_mlp<<<gridFor(128 * 128), 256, 0, stream>>>(upd_w2, 128, 128, 128, in_gamma, sw_u2);
    pg_fold_bias<<<1, 128, 0, stream>>>(upd_b2, in_gamma, in_beta, b_fold);
    for (int i = 0; i < 3; ++i) {
        pg_swizzle_conv<<<gridFor(27 * 128 * 128), 256, 0, stream>>>(
            blk_conv1 + (size_t)i * 128 * 128 * 27, 128, 128, 27, blk_bn1_g + i * 128, sw_c1[i]);
        pg_swizzle_conv<<<gridFor(27 * 128 * 128), 256, 0, stream>>>(
            blk_conv2 + (size_t)i * 128 * 128 * 27, 128, 128, 27, blk_bn2_g + i * 128, sw_c2[i]);
        pg_swizzle_conv<<<gridFor(125 * 128 * 128), 256, 0, stream>>>(
            blk_conv5 + (size_t)i * 128 * 128 * 125, 128, 128, 125, blk_bns_g + i * 128, sw_c5[i]);
    }

    // --- MPNN: fused edge pipeline + scatter-mean, then update MLP ---
    pg_edge_kernel<<<4096, 256, 0, stream>>>(node_emb, node_pos, grid_pos, edge_row, edge_col,
                                             sw_e1, edge_b1, sw_e2, edge_b2,
                                             sw_m1, msg_b1, sw_m2, msg_b2, aggbuf);
    pg_update_kernel<<<512, 256, 0, stream>>>(aggbuf, sw_u1, upd_b1, sw_u2, b_fold, act_a);

    // --- 3 residual conv blocks (ping-pong), 128 blocks: (b, z, y-tile of 4) ---
    const _Float16* xc = act_a;
    _Float16* xn = act_b;
    for (int i = 0; i < 3; ++i) {
        pg_conv_kernel<3, true, false><<<128, 256, 0, stream>>>(xc, sw_c1[i], blk_bn1_b + i * 128,
                                                                nullptr, tmp1);
        pg_conv_kernel<3, false, false><<<128, 256, 0, stream>>>(tmp1, sw_c2[i], blk_bn2_b + i * 128,
                                                                 nullptr, tmp2);
        pg_conv_kernel<5, true, true><<<128, 256, 0, stream>>>(xc, sw_c5[i], blk_bns_b + i * 128,
                                                               tmp2, xn);
        const _Float16* t = xc; xc = xn; xn = (_Float16*)t;
    }

    // --- pool + fc + log_softmax ---
    pg_pool_kernel<<<2, 256, 0, stream>>>(xc, pooled);
    pg_head_kernel<<<1, 64, 0, stream>>>(pooled, fc_w, fc_b, out);
    (void)in_sizes; (void)n_in; (void)out_size; (void)ws_size;
}
